// DoubleRelaxedContrastiveLoss_76270029243000
// MI455X (gfx1250) — matrix operand order repaired and checked by reading, not compiled, not run on hardware
//
#include <hip/hip_runtime.h>

typedef __attribute__((ext_vector_type(16))) _Float16 v16h;
typedef __attribute__((ext_vector_type(8)))  _Float16 v8h;
typedef __attribute__((ext_vector_type(8)))  float    v8f;

#define B_SZ 8192
#define D_SZ 256
#define ALPHA_C      0.7f
#define LAMBDA_NEG_C 0.7f
#define GAMMA_SEMI_C 0.7f
#define LDST 40   // LDS row stride in halves (32 data + 8 pad -> bank spread)

// ---------------------------------------------------------------------------
// Kernel 1: L2-normalize rows, cast to f16. One wave per row, 8 rows/block.
// blockIdx.y: 0 = text, 1 = image.
// ---------------------------------------------------------------------------
__global__ void __launch_bounds__(256)
nrm_kernel(const float* __restrict__ t, const float* __restrict__ v,
           _Float16* __restrict__ tn, _Float16* __restrict__ vn) {
    const float* src = (blockIdx.y == 0) ? t : v;
    _Float16*    dst = (blockIdx.y == 0) ? tn : vn;
    int wave = threadIdx.x >> 5, lane = threadIdx.x & 31;
    int row  = blockIdx.x * 8 + wave;
    const float* p = src + (size_t)row * D_SZ + lane * 8;
    float4 a = *(const float4*)p;
    float4 b = *(const float4*)(p + 4);
    float ss = a.x*a.x + a.y*a.y + a.z*a.z + a.w*a.w
             + b.x*b.x + b.y*b.y + b.z*b.z + b.w*b.w;
    for (int o = 16; o > 0; o >>= 1) ss += __shfl_xor(ss, o, 32);
    float scale = 1.0f / fmaxf(sqrtf(ss), 1e-12f);
    v8h h;
    h[0] = (_Float16)(a.x * scale); h[1] = (_Float16)(a.y * scale);
    h[2] = (_Float16)(a.z * scale); h[3] = (_Float16)(a.w * scale);
    h[4] = (_Float16)(b.x * scale); h[5] = (_Float16)(b.y * scale);
    h[6] = (_Float16)(b.z * scale); h[7] = (_Float16)(b.w * scale);
    *(v8h*)(dst + (size_t)row * D_SZ + lane * 8) = h;
}

// ---------------------------------------------------------------------------
// Kernel 2: fused sim = tn @ vn^T with loss epilogue (generic + diagonal).
// Block tile 256x128, 8 waves in 4x2 grid, each wave 64x64 = 4x4 WMMA tiles.
// 16 ds_load_b128 : 16 v_wmma per wave per K-step (1:1 compute density).
// ---------------------------------------------------------------------------
__global__ void __launch_bounds__(256)
gemm_kernel(const _Float16* __restrict__ tmat, const _Float16* __restrict__ vmat,
            float* __restrict__ partials) {
    __shared__ _Float16 sA[256 * LDST];
    __shared__ _Float16 sB[128 * LDST];
    __shared__ float wsum[8];

    const int tid  = threadIdx.x;
    const int wave = tid >> 5, lane = tid & 31;
    const int l  = lane & 15, hi = lane >> 4;     // half-wave split per ISA layout
    const int wr = wave >> 1, wc = wave & 1;      // wave tile: rows wr*64, cols wc*64

    const int rowBase = blockIdx.y * 256;
    const int colBase = blockIdx.x * 128;

    v8f acc[4][4] = {};

    // staging per K-step:
    //  A: 256 rows x 32 halves, thread t -> row t, full 32-half slab (4x16B)
    //  B: 128 rows x 32 halves, thread t -> row t/2, 16-half segment (2x16B)
    const _Float16* gA = tmat + (size_t)(rowBase + tid) * D_SZ;
    const int bRow = tid >> 1;
    const int bSeg = (tid & 1) * 16;
    const _Float16* gB = vmat + (size_t)(colBase + bRow) * D_SZ + bSeg;
    _Float16* lA = sA + tid  * LDST;
    _Float16* lB = sB + bRow * LDST + bSeg;

    for (int k0 = 0; k0 < D_SZ; k0 += 32) {
        __syncthreads();                       // previous iter consumers done
        *(uint4*)(lA)      = *(const uint4*)(gA + k0);
        *(uint4*)(lA + 8)  = *(const uint4*)(gA + k0 + 8);
        *(uint4*)(lA + 16) = *(const uint4*)(gA + k0 + 16);
        *(uint4*)(lA + 24) = *(const uint4*)(gA + k0 + 24);
        *(uint4*)(lB)      = *(const uint4*)(gB + k0);
        *(uint4*)(lB + 8)  = *(const uint4*)(gB + k0 + 8);
        __syncthreads();

        // A fragments: lane l (hi=0) holds K{0..7,16..23}; hi=1 -> K{8..15,24..31}
        union { v16h v; v8h h[2]; } af[4], bf[4];
#pragma unroll
        for (int tm = 0; tm < 4; ++tm) {
            int r = wr * 64 + tm * 16 + l;
            af[tm].h[0] = *(const v8h*)&sA[r * LDST + hi * 8];
            af[tm].h[1] = *(const v8h*)&sA[r * LDST + 16 + hi * 8];
        }
        // B fragments: lane l holds vn row (col of sim), hi=0 -> K{0..7,16..23}; hi=1 -> K{8..15,24..31}
#pragma unroll
        for (int tc = 0; tc < 4; ++tc) {
            int r = wc * 64 + tc * 16 + l;
            bf[tc].h[0] = *(const v8h*)&sB[r * LDST + hi * 8];
            bf[tc].h[1] = *(const v8h*)&sB[r * LDST + 16 + hi * 8];
        }
#pragma unroll
        for (int tm = 0; tm < 4; ++tm)
#pragma unroll
            for (int tc = 0; tc < 4; ++tc)
                acc[tm][tc] = __builtin_amdgcn_wmma_f32_16x16x32_f16(
                    false, af[tm].v, false, bf[tc].v,
                    (short)0, acc[tm][tc], false, false);
    }

    // Epilogue: C/D layout -> VGPR r: lanes 0-15 M=r, lanes 16-31 M=r+8, N=l
    float p = 0.0f;
#pragma unroll
    for (int tm = 0; tm < 4; ++tm)
#pragma unroll
        for (int tc = 0; tc < 4; ++tc)
#pragma unroll
            for (int r = 0; r < 8; ++r) {
                int i = rowBase + wr * 64 + tm * 16 + hi * 8 + r;
                int j = colBase + wc * 64 + tc * 16 + l;
                float s  = acc[tm][tc][r];
                float rl = fmaxf(s, 0.0f);
                p += (i == j) ? (s - 1.0f) * (s - 1.0f)
                              : LAMBDA_NEG_C * rl * rl;
            }
    for (int o = 16; o > 0; o >>= 1) p += __shfl_xor(p, o, 32);
    if (lane == 0) wsum[wave] = p;
    __syncthreads();
    if (tid == 0) {
        float s = 0.0f;
        for (int w = 0; w < 8; ++w) s += wsum[w];
        partials[blockIdx.y * gridDim.x + blockIdx.x] = s;
    }
}

// ---------------------------------------------------------------------------
// Kernel 3: semi-positive corrections. One wave per row; <=3 dots of dim 256.
// Adds gamma*relu(alpha-s)^2 - lambda*relu(s)^2 for each unique valid index.
// ---------------------------------------------------------------------------
__global__ void __launch_bounds__(256)
semi_kernel(const _Float16* __restrict__ tmat, const _Float16* __restrict__ vmat,
            const int* __restrict__ idx, float* __restrict__ partials) {
    __shared__ float wsum[8];
    int wave = threadIdx.x >> 5, lane = threadIdx.x & 31;
    int i = blockIdx.x * 8 + wave;

    v8h th = *(const v8h*)(tmat + (size_t)i * D_SZ + lane * 8);
    float tf[8];
#pragma unroll
    for (int c = 0; c < 8; ++c) tf[c] = (float)th[c];

    int j0 = idx[i * 3 + 0], j1 = idx[i * 3 + 1], j2 = idx[i * 3 + 2];
    int  js[3] = { j0, j1, j2 };
    bool vs[3] = { (j0 >= 0) && (j0 != i),
                   (j1 >= 0) && (j1 != i) && (j1 != j0),
                   (j2 >= 0) && (j2 != i) && (j2 != j0) && (j2 != j1) };

    float acc = 0.0f;
    for (int k = 0; k < 3; ++k) {           // wave-uniform branches (i uniform)
        if (!vs[k]) continue;
        int j = js[k];
        v8h vh = *(const v8h*)(vmat + (size_t)j * D_SZ + lane * 8);
        float d = 0.0f;
#pragma unroll
        for (int c = 0; c < 8; ++c) d = fmaf(tf[c], (float)vh[c], d);
        for (int o = 16; o > 0; o >>= 1) d += __shfl_xor(d, o, 32);
        float rn = fmaxf(ALPHA_C - d, 0.0f);
        float rp = fmaxf(d, 0.0f);
        acc += GAMMA_SEMI_C * rn * rn - LAMBDA_NEG_C * rp * rp;
    }
    if (lane == 0) wsum[wave] = acc;
    __syncthreads();
    if (threadIdx.x == 0) {
        float s = 0.0f;
        for (int w = 0; w < 8; ++w) s += wsum[w];
        partials[blockIdx.x] = s;
    }
}

// ---------------------------------------------------------------------------
// Kernel 4: deterministic final reduction of 2048 + 1024 partials.
// ---------------------------------------------------------------------------
__global__ void __launch_bounds__(1024)
reduce_kernel(const float* __restrict__ partials, float* __restrict__ out) {
    __shared__ float sh[1024];
    int t = threadIdx.x;
    float s = 0.0f;
    for (int k = t; k < 3072; k += 1024) s += partials[k];
    sh[t] = s;
    __syncthreads();
    for (int o = 512; o > 0; o >>= 1) {
        if (t < o) sh[t] += sh[t + o];
        __syncthreads();
    }
    if (t == 0) out[0] = sh[0];
}

// ---------------------------------------------------------------------------
extern "C" void kernel_launch(void* const* d_in, const int* in_sizes, int n_in,
                              void* d_out, int out_size, void* d_ws, size_t ws_size,
                              hipStream_t stream) {
    (void)in_sizes; (void)n_in; (void)out_size; (void)ws_size;
    const float* text  = (const float*)d_in[0];
    const float* image = (const float*)d_in[1];
    const int*   idx   = (const int*)d_in[2];

    _Float16* tn = (_Float16*)d_ws;
    _Float16* vn = tn + (size_t)B_SZ * D_SZ;
    float* partials = (float*)((char*)d_ws + (size_t)2 * B_SZ * D_SZ * sizeof(_Float16));
    float* out = (float*)d_out;

    nrm_kernel   <<<dim3(B_SZ / 8, 2), 256, 0, stream>>>(text, image, tn, vn);
    gemm_kernel  <<<dim3(64, 32),      256, 0, stream>>>(tn, vn, partials);
    semi_kernel  <<<B_SZ / 8,          256, 0, stream>>>(tn, vn, idx, partials + 2048);
    reduce_kernel<<<1,                1024, 0, stream>>>(partials, out);
}
